// Gate_27195732918640
// MI455X (gfx1250) — compile-verified, matrix operand order checked by reading, not run on hardware
//
#include <hip/hip_runtime.h>
#include <hip/hip_bf16.h>
#include <stdint.h>

// ---- problem constants (match reference) ----
#define NTOK   16384
#define DIM    7168
#define NEXP   256
#define NGRP   8
#define GSIZE  32          // experts per group
#define TOPKG  4
#define TOPK   8
#define ROUTE_SCALE 2.5f
#define CLAMP_MIN   1e-7f

#define KSTEP  32          // WMMA bf16 K per step
#define BM     128         // tokens per block (8 waves x 16)
#define W_ELEMS (NEXP * DIM)

typedef __attribute__((ext_vector_type(16))) __bf16 v16bf;
typedef __attribute__((ext_vector_type(8)))  float  v8f;

// -------- bf16 split helpers (round-to-nearest-even) --------
__device__ __forceinline__ unsigned short f2bf_rne(float f) {
  unsigned int u = __float_as_uint(f);
  u += 0x7FFFu + ((u >> 16) & 1u);
  return (unsigned short)(u >> 16);
}
__device__ __forceinline__ float bf2f(unsigned short h) {
  return __uint_as_float(((unsigned int)h) << 16);
}

// -------- optional CDNA5 async global->LDS path --------
#if defined(__has_builtin)
#if __has_builtin(__builtin_amdgcn_global_load_async_to_lds_b128)
#define HAS_ASYNC_LDS 1
#endif
#endif

#if defined(HAS_ASYNC_LDS)
// Matches the builtin's parameter types from the compiler diagnostic:
//   (int __vector_size(16) AS1*, int __vector_size(16) AS3*, Imm off, Imm cpol)
typedef int v4i_vs __attribute__((vector_size(16)));
typedef __attribute__((address_space(1))) v4i_vs gv4i_t;
typedef __attribute__((address_space(3))) v4i_vs lv4i_t;
#endif

__device__ __forceinline__ void wait_async_zero() {
#if defined(__has_builtin)
#if __has_builtin(__builtin_amdgcn_s_wait_asynccnt)
  __builtin_amdgcn_s_wait_asynccnt(0);
#else
  asm volatile("s_wait_asynccnt 0" ::: "memory");
#endif
#else
  asm volatile("s_wait_asynccnt 0" ::: "memory");
#endif
}

// =====================================================================
// Kernel 1: split fp32 weight [E][D] into bf16 hi/lo in K-blocked layout
//   whi/wlo[(k/32)*E*32 + e*32 + (k%32)]
// =====================================================================
__global__ __launch_bounds__(256) void convert_weight_kernel(
    const float* __restrict__ w,
    unsigned short* __restrict__ whi,
    unsigned short* __restrict__ wlo) {
  int i = blockIdx.x * 256 + threadIdx.x;
  if (i >= W_ELEMS) return;
  int e = i / DIM;
  int k = i - e * DIM;
  float f = w[i];
  unsigned short h = f2bf_rne(f);
  unsigned short l = f2bf_rne(f - bf2f(h));
  size_t o = (size_t)(k >> 5) * (NEXP * KSTEP) + (size_t)e * KSTEP + (k & 31);
  whi[o] = h;
  wlo[o] = l;
}

// =====================================================================
// Kernel 2: scores[n][e] = sigmoid(x[n,:] . w[e,:])  via bf16x3 WMMA
//   block: 256 threads = 8 waves; wave w -> tokens [blk*128 + w*16, +16),
//   all 256 experts (16 N-tiles). Weight K-chunk staged in LDS per step.
// =====================================================================
__global__ __launch_bounds__(256) void gate_gemm_kernel(
    const float* __restrict__ x,
    const unsigned short* __restrict__ whi,
    const unsigned short* __restrict__ wlo,
    float* __restrict__ scores) {
  __shared__ __align__(32) unsigned short sWhi[NEXP * KSTEP];  // 16 KB
  __shared__ __align__(32) unsigned short sWlo[NEXP * KSTEP];  // 16 KB

  const int tid  = threadIdx.x;
  const int lane = tid & 31;
  const int wv   = tid >> 5;
  const int kh   = lane >> 4;   // K-half selector (0/1)
  const int nl   = lane & 15;   // row-in-tile index
  const int tokbase = blockIdx.x * BM + wv * 16;

  const float* xrow = x + (size_t)(tokbase + nl) * DIM + kh * 16;

  v8f acc[16];
#pragma unroll
  for (int t = 0; t < 16; ++t) acc[t] = (v8f){0, 0, 0, 0, 0, 0, 0, 0};

  for (int k0 = 0; k0 < DIM; k0 += KSTEP) {
    // ---- stage weight chunk (hi+lo, 32 KB) into LDS: thread tid owns row tid ----
    {
      const size_t cb = (size_t)(k0 >> 5) * (NEXP * KSTEP);
      const unsigned short* sh = whi + cb + (size_t)tid * KSTEP;
      const unsigned short* sl = wlo + cb + (size_t)tid * KSTEP;
      unsigned short* dh = &sWhi[tid * KSTEP];
      unsigned short* dl = &sWlo[tid * KSTEP];
#if defined(HAS_ASYNC_LDS)
#pragma unroll
      for (int q = 0; q < 4; ++q) {
        __builtin_amdgcn_global_load_async_to_lds_b128(
            (gv4i_t*)(sh + q * 8), (lv4i_t*)(dh + q * 8), 0, 0);
        __builtin_amdgcn_global_load_async_to_lds_b128(
            (gv4i_t*)(sl + q * 8), (lv4i_t*)(dl + q * 8), 0, 0);
      }
      wait_async_zero();
#else
#pragma unroll
      for (int q = 0; q < 4; ++q) {
        ((uint4*)dh)[q] = ((const uint4*)sh)[q];
        ((uint4*)dl)[q] = ((const uint4*)sl)[q];
      }
#endif
    }
    __syncthreads();

    // ---- A fragment: 16 contiguous fp32 per lane -> bf16 hi/lo ----
    float a[16];
    const float* xp = xrow + k0;
#pragma unroll
    for (int q = 0; q < 4; ++q) ((float4*)a)[q] = ((const float4*)xp)[q];

    union { v16bf v; unsigned short s[16]; } ahi, alo;
#pragma unroll
    for (int i = 0; i < 16; ++i) {
      unsigned short h = f2bf_rne(a[i]);
      ahi.s[i] = h;
      alo.s[i] = f2bf_rne(a[i] - bf2f(h));
    }

    // ---- 16 expert tiles, 3 WMMAs each (hi*hi + hi*lo + lo*hi) ----
#pragma unroll
    for (int t = 0; t < 16; ++t) {
      const int e = t * 16 + nl;  // expert row in chunk
      const v16bf bhi = *(const v16bf*)&sWhi[e * KSTEP + kh * 16];
      const v16bf blo = *(const v16bf*)&sWlo[e * KSTEP + kh * 16];
      acc[t] = __builtin_amdgcn_wmma_f32_16x16x32_bf16(
          false, ahi.v, false, bhi, (short)0, acc[t], false, false);
      acc[t] = __builtin_amdgcn_wmma_f32_16x16x32_bf16(
          false, ahi.v, false, blo, (short)0, acc[t], false, false);
      acc[t] = __builtin_amdgcn_wmma_f32_16x16x32_bf16(
          false, alo.v, false, bhi, (short)0, acc[t], false, false);
    }
    __syncthreads();
  }

  // ---- epilogue: sigmoid + store (C layout: vgpr j, half kh -> token j+8*kh) ----
#pragma unroll
  for (int t = 0; t < 16; ++t) {
#pragma unroll
    for (int j = 0; j < 8; ++j) {
      float v  = acc[t][j];
      float sg = 1.0f / (1.0f + __expf(-v));
      int token = tokbase + j + 8 * kh;
      scores[(size_t)token * NEXP + t * 16 + nl] = sg;
    }
  }
}

// =====================================================================
// Kernel 3: routing. One thread per token (L2-resident score rows).
// =====================================================================
__global__ __launch_bounds__(256) void routing_kernel(
    const float* __restrict__ scores,
    const float* __restrict__ bias,
    float* __restrict__ out_w,
    float* __restrict__ out_i) {
  __shared__ float sbias[NEXP];
  sbias[threadIdx.x] = bias[threadIdx.x];
  __syncthreads();

  const int token = blockIdx.x * 256 + threadIdx.x;
  const float* row = scores + (size_t)token * NEXP;

  // pass 1: group score = sum of top-2 biased scores per 32-expert group
  float gsc[NGRP];
#pragma unroll
  for (int g = 0; g < NGRP; ++g) {
    float m1 = -3.0e38f, m2 = -3.0e38f;
    for (int j = 0; j < GSIZE; ++j) {
      float v = row[g * GSIZE + j] + sbias[g * GSIZE + j];
      if (v > m1) { m2 = m1; m1 = v; }
      else if (v > m2) { m2 = v; }
    }
    gsc[g] = m1 + m2;
  }

  // top-4 groups (lowest index wins ties, like jax top_k)
  unsigned gmask = 0;
#pragma unroll
  for (int r = 0; r < TOPKG; ++r) {
    int best = 0;
    float bv = -3.0e38f;
#pragma unroll
    for (int g = 0; g < NGRP; ++g) {
      if (!((gmask >> g) & 1u) && gsc[g] > bv) { bv = gsc[g]; best = g; }
    }
    gmask |= (1u << best);
  }

  // pass 2: stable top-8 over masked biased scores (out-group -> 0.0, as ref)
  float topv[TOPK], topw[TOPK];
  int   topi[TOPK];
#pragma unroll
  for (int k = 0; k < TOPK; ++k) { topv[k] = -3.0e38f; topw[k] = 0.f; topi[k] = 0; }

  for (int e = 0; e < NEXP; ++e) {
    float orig = row[e];
    float v = ((gmask >> (e >> 5)) & 1u) ? (orig + sbias[e]) : 0.0f;
    if (v > topv[TOPK - 1]) {
      bool done = false;
#pragma unroll
      for (int j = TOPK - 1; j >= 1; --j) {
        if (!done) {
          if (v > topv[j - 1]) {
            topv[j] = topv[j - 1]; topw[j] = topw[j - 1]; topi[j] = topi[j - 1];
          } else {
            topv[j] = v; topw[j] = orig; topi[j] = e; done = true;
          }
        }
      }
      if (!done) { topv[0] = v; topw[0] = orig; topi[0] = e; }
    }
  }

  // clamp, normalize, scale; indices emitted as exact float values
  float sum = 0.f;
  float cw[TOPK];
#pragma unroll
  for (int k = 0; k < TOPK; ++k) {
    cw[k] = topw[k] > CLAMP_MIN ? topw[k] : CLAMP_MIN;
    sum += cw[k];
  }
  float inv = ROUTE_SCALE / sum;
#pragma unroll
  for (int k = 0; k < TOPK; ++k) {
    out_w[(size_t)token * TOPK + k] = cw[k] * inv;
    out_i[(size_t)token * TOPK + k] = (float)topi[k];
  }
}

// =====================================================================
extern "C" void kernel_launch(void* const* d_in, const int* in_sizes, int n_in,
                              void* d_out, int out_size, void* d_ws, size_t ws_size,
                              hipStream_t stream) {
  (void)in_sizes; (void)n_in; (void)out_size; (void)ws_size;
  const float* x    = (const float*)d_in[0];   // [16384, 7168]
  const float* w    = (const float*)d_in[1];   // [256, 7168]
  const float* bias = (const float*)d_in[2];   // [256]

  // workspace layout: whi (bf16), wlo (bf16), scores (fp32)  ~24.1 MB total
  unsigned short* whi = (unsigned short*)d_ws;
  unsigned short* wlo = whi + (size_t)W_ELEMS;
  float* scores = (float*)(wlo + (size_t)W_ELEMS);

  float* out_w = (float*)d_out;                        // [16384, 8] weights
  float* out_i = out_w + (size_t)NTOK * TOPK;          // [16384, 8] indices (as float)

  convert_weight_kernel<<<(W_ELEMS + 255) / 256, 256, 0, stream>>>(w, whi, wlo);
  gate_gemm_kernel<<<NTOK / BM, 256, 0, stream>>>(x, whi, wlo, scores);
  routing_kernel<<<NTOK / 256, 256, 0, stream>>>(scores, bias, out_w, out_i);
}
